// GRU_Model_14731737825928
// MI455X (gfx1250) — matrix operand order repaired and checked by reading, not compiled, not run on hardware
//
#include <hip/hip_runtime.h>
#include <hip/hip_bf16.h>
#include <cmath>

// Problem constants
constexpr int Bx = 512;    // batch
constexpr int Tt = 64;     // time steps
constexpr int Ee = 300;    // embedding dim
constexpr int Ep = 320;    // padded embedding dim (multiple of 32)
constexpr int Hh = 512;    // hidden
constexpr int G3 = 1536;   // 3*H (gates per direction)
constexpr int NG = 3072;   // 2*3*H (both directions)
constexpr int Mm = Bx * Tt; // 32768 rows

typedef __bf16 bf16;
typedef __attribute__((ext_vector_type(16))) __bf16 v16bf;
typedef __attribute__((ext_vector_type(8)))  __bf16 v8bf;
typedef __attribute__((ext_vector_type(8)))  float  v8f;
typedef __attribute__((ext_vector_type(4)))  unsigned int u32x4;
typedef __attribute__((ext_vector_type(8)))  int i32x8;
typedef __attribute__((ext_vector_type(4)))  int i32x4;

static __device__ inline v8f wmma_bf16(v16bf a, v16bf b, v8f c) {
  // D = A(16x32 bf16) x B(32x16 bf16) + C(16x16 f32)
  return __builtin_amdgcn_wmma_f32_16x16x32_bf16(false, a, false, b,
                                                 (short)0, c, false, false);
}

// A-matrix fragment (16x32 bf16), per ISA 7.12.2:
// lane L: m = L&15, h = L>>4; elems 0..7 -> K=8h..8h+7, elems 8..15 -> K=16+8h..
static __device__ inline v16bf load_a_frag(const bf16* __restrict__ base, int ld, int k0) {
  int lane = threadIdx.x & 31;
  int m = lane & 15, h = lane >> 4;
  const bf16* p = base + (size_t)m * ld + k0 + 8 * h;
  v8bf lo = *(const v8bf*)p;
  v8bf hi = *(const v8bf*)(p + 16);
  v16bf r;
#pragma unroll
  for (int i = 0; i < 8; ++i) { r[i] = lo[i]; r[8 + i] = hi[i]; }
  return r;
}

// B-matrix fragment (32x16 bf16). Weights row-major (N,K) == B^T in memory:
// lane L: n = L&15, h = L>>4; elems 0..7 -> K=8h..8h+7, elems 8..15 -> K=16+8h..
static __device__ inline v16bf load_b_frag(const bf16* __restrict__ base, int ld, int k0) {
  int lane = threadIdx.x & 31;
  int n = lane & 15, h = lane >> 4;
  const bf16* p = base + (size_t)n * ld + k0 + 8 * h;
  v8bf lo = *(const v8bf*)p;
  v8bf hi = *(const v8bf*)(p + 16);
  v16bf r;
#pragma unroll
  for (int i = 0; i < 8; ++i) { r[i] = lo[i]; r[8 + i] = hi[i]; }
  return r;
}

// Tensor Data Mover: DMA a 2D tile (rows x cols bf16, row stride = stride_elems)
// from global memory into LDS. Issued once per calling wave; tracked by TENSORcnt.
// D# layout per CDNA5 ISA. 6-arg builtin (clang-23 lane).
static __device__ inline void tdm_load_tile(void* lds, const bf16* g,
                                            int stride_elems, int rows, int cols) {
  unsigned lds_off = (unsigned)(size_t)lds;            // flat addr[31:0] == LDS offset
  unsigned long long ga = (unsigned long long)(size_t)g;
  u32x4 g0;
  g0[0] = 1u;                                          // count=1 (valid), user mode
  g0[1] = lds_off;                                     // lds_addr
  g0[2] = (unsigned)(ga & 0xFFFFFFFFu);                // global_addr[31:0]
  g0[3] = (unsigned)((ga >> 32) & 0x1FFFFFFu) | (2u << 30); // addr[56:32], type=2
  unsigned td0 = (unsigned)cols;                       // tensor_dim0 == tile_dim0
  unsigned td1 = (unsigned)rows;                       // tensor_dim1 == tile_dim1
  unsigned long long s0 = (unsigned long long)stride_elems; // dim0 stride (elems)
  i32x8 g1;
  g1[0] = (int)(1u << 16);                             // data_size=1 (2 bytes)
  g1[1] = (int)((td0 & 0xFFFFu) << 16);                // tensor_dim0[15:0]
  g1[2] = (int)(((td0 >> 16) & 0xFFFFu) | ((td1 & 0xFFFFu) << 16));
  g1[3] = (int)(((td1 >> 16) & 0xFFFFu) | (((unsigned)cols & 0xFFFFu) << 16)); // tile_dim0
  g1[4] = (int)((unsigned)rows & 0xFFFFu);             // tile_dim1; tile_dim2=0
  g1[5] = (int)(s0 & 0xFFFFFFFFu);                     // tensor_dim0_stride[31:0]
  g1[6] = (int)((s0 >> 32) & 0xFFFFu);                 // stride[47:32]; dim1_stride=0
  g1[7] = 0;
  i32x4 gz4 = {0, 0, 0, 0};
  i32x8 gz8 = {0, 0, 0, 0, 0, 0, 0, 0};
  __builtin_amdgcn_tensor_load_to_lds(g0, g1, gz4, gz4, gz8, 0);
}

// fp32 (rows x K) -> bf16 (rows x Kp), zero-padded tail
__global__ void k_cast_pad(const float* __restrict__ src, bf16* __restrict__ dst,
                           int K, int Kp) {
  int row = blockIdx.x;
  for (int c = threadIdx.x; c < Kp; c += blockDim.x)
    dst[(size_t)row * Kp + c] = (c < K) ? (bf16)src[(size_t)row * K + c] : (bf16)0.0f;
}

// Embedding gather: x[b*T+t][0..319] = bf16(emb[token]), zero pad cols 300..319
__global__ void k_embed(const int* __restrict__ tokens, const float* __restrict__ emb,
                        bf16* __restrict__ x) {
  int row = blockIdx.x;               // b*T + t
  int tok = tokens[row];
  const float* e = emb + (size_t)tok * Ee;
  for (int c = threadIdx.x; c < Ep; c += blockDim.x)
    x[(size_t)row * Ep + c] = (c < Ee) ? (bf16)e[c] : (bf16)0.0f;
}

// D(M,N bf16) = A(M,K bf16) @ W(N,K bf16)^T, TDM-staged through LDS, double buffered.
// Block: 256 thr = 8 waves (2M x 4N), wave tile 32x32 (2x2 wmma), block tile 64x128.
constexpr int BM = 64, BN = 128, BK = 64;
__global__ __launch_bounds__(256)
void k_gemm_bf16(const bf16* __restrict__ A, const bf16* __restrict__ W,
                 bf16* __restrict__ D, int K, int ldd) {
  __shared__ bf16 sA[2][BM * BK];   // 2 x 8 KB
  __shared__ bf16 sB[2][BN * BK];   // 2 x 16 KB
  int w  = threadIdx.x >> 5;
  int wm = w >> 2;                  // 0..1
  int wn = w & 3;                   // 0..3
  int m0 = blockIdx.x * BM;
  int n0 = blockIdx.y * BN;
  const v8f vzero = {0.f,0.f,0.f,0.f,0.f,0.f,0.f,0.f};
  v8f acc[2][2];
#pragma unroll
  for (int i = 0; i < 2; ++i)
#pragma unroll
    for (int j = 0; j < 2; ++j) acc[i][j] = vzero;

  int nch = K / BK;
  if (w == 0) {   // wave 0 drives the DMA pipeline (EXEC-independent, once per wave)
    tdm_load_tile(&sA[0][0], A + (size_t)m0 * K, K, BM, BK);
    tdm_load_tile(&sB[0][0], W + (size_t)n0 * K, K, BN, BK);
  }
  for (int c = 0; c < nch; ++c) {
    int cur = c & 1;
    if (w == 0) __builtin_amdgcn_s_wait_tensorcnt(0);  // chunk c landed in LDS
    __syncthreads();              // publish buf[cur]; proves buf[cur^1] fully consumed
    if (w == 0 && c + 1 < nch) {
      tdm_load_tile(&sA[cur ^ 1][0], A + (size_t)m0 * K + (c + 1) * BK, K, BM, BK);
      tdm_load_tile(&sB[cur ^ 1][0], W + (size_t)n0 * K + (c + 1) * BK, K, BN, BK);
    }
    const bf16* tA = &sA[cur][0] + (wm * 32) * BK;
    const bf16* tB = &sB[cur][0] + (wn * 32) * BK;
#pragma unroll
    for (int kk = 0; kk < BK; kk += 32) {
      v16bf a0 = load_a_frag(tA, BK, kk);
      v16bf a1 = load_a_frag(tA + 16 * BK, BK, kk);
      v16bf b0 = load_b_frag(tB, BK, kk);
      v16bf b1 = load_b_frag(tB + 16 * BK, BK, kk);
      acc[0][0] = wmma_bf16(a0, b0, acc[0][0]);
      acc[0][1] = wmma_bf16(a0, b1, acc[0][1]);
      acc[1][0] = wmma_bf16(a1, b0, acc[1][0]);
      acc[1][1] = wmma_bf16(a1, b1, acc[1][1]);
    }
  }
  // C/D layout: VGPR r, lane L -> m = r + 8*(L>>4), n = L&15
  int lane = threadIdx.x & 31;
  int hh = lane >> 4, nn = lane & 15;
#pragma unroll
  for (int i = 0; i < 2; ++i)
#pragma unroll
    for (int j = 0; j < 2; ++j) {
      bf16* drow = D + (size_t)(m0 + wm * 32 + i * 16 + 8 * hh) * ldd
                     + (n0 + wn * 32 + j * 16 + nn);
#pragma unroll
      for (int r = 0; r < 8; ++r)
        drow[(size_t)r * ldd] = (bf16)acc[i][j][r];
    }
}

// One GRU time step (one layer), fused: gh = h_prev @ Whh^T for all 3 gates via WMMA,
// then gate math in fp32. dir = dirbase + blockIdx.z. Ping-pong state in/out.
__global__ __launch_bounds__(256)
void k_gru_step(const bf16* __restrict__ hb_in, const float* __restrict__ hf_in,
                const bf16* __restrict__ whh,   const bf16* __restrict__ gx,
                const float* __restrict__ bih,  const float* __restrict__ bhh,
                float* __restrict__ hf_out, bf16* __restrict__ hb_out,
                bf16* __restrict__ seq_out, float* __restrict__ last_out,
                int s, int dirbase, int write_last) {
  int dir = dirbase + blockIdx.z;
  int w   = threadIdx.x >> 5;
  int wm  = w & 1;                  // 0..1 (batch)
  int wn  = w >> 1;                 // 0..3 (hidden)
  int m0  = blockIdx.x * 32 + wm * 16;   // batch tile
  int n0  = blockIdx.y * 64 + wn * 16;   // hidden-unit tile
  const v8f vzero = {0.f,0.f,0.f,0.f,0.f,0.f,0.f,0.f};
  v8f cr = vzero, cz = vzero, cn = vzero;

  const bf16* Ab = hb_in + ((size_t)dir * Bx + m0) * Hh;
  const bf16* Wr = whh + ((size_t)dir * G3 + 0 * Hh + n0) * Hh;
  const bf16* Wz = whh + ((size_t)dir * G3 + 1 * Hh + n0) * Hh;
  const bf16* Wn = whh + ((size_t)dir * G3 + 2 * Hh + n0) * Hh;
  for (int k0 = 0; k0 < Hh; k0 += 32) {
    v16bf a = load_a_frag(Ab, Hh, k0);
    cr = wmma_bf16(a, load_b_frag(Wr, Hh, k0), cr);
    cz = wmma_bf16(a, load_b_frag(Wz, Hh, k0), cz);
    cn = wmma_bf16(a, load_b_frag(Wn, Hh, k0), cn);
  }

  int lane = threadIdx.x & 31;
  int hh2 = lane >> 4, nn = lane & 15;
  int n = n0 + nn;
  int tmap = (dir == 0) ? s : (Tt - 1 - s);  // fwd t=s; bwd t=T-1-s
#pragma unroll
  for (int i = 0; i < 8; ++i) {
    int b = m0 + i + 8 * hh2;
    size_t grow = ((size_t)b * Tt + tmap) * NG + (size_t)dir * G3;
    float xr = (float)gx[grow + n]          + bih[dir * G3 + n];
    float xz = (float)gx[grow + Hh + n]     + bih[dir * G3 + Hh + n];
    float xn = (float)gx[grow + 2 * Hh + n] + bih[dir * G3 + 2 * Hh + n];
    float hr = cr[i] + bhh[dir * G3 + n];
    float hz = cz[i] + bhh[dir * G3 + Hh + n];
    float hn = cn[i] + bhh[dir * G3 + 2 * Hh + n];
    float r  = 1.f / (1.f + __expf(-(xr + hr)));
    float z  = 1.f / (1.f + __expf(-(xz + hz)));
    float nv = tanhf(xn + r * hn);
    float hp = hf_in[((size_t)dir * Bx + b) * Hh + n];
    float hnew = (1.f - z) * nv + z * hp;
    hf_out[((size_t)dir * Bx + b) * Hh + n] = hnew;
    hb_out[((size_t)dir * Bx + b) * Hh + n] = (bf16)hnew;
    if (seq_out)
      seq_out[((size_t)b * Tt + tmap) * (2 * Hh) + dir * Hh + n] = (bf16)hnew;
    if (write_last)
      last_out[(size_t)b * (2 * Hh) + dir * Hh + n] = hnew;
  }
}

// Tiny final FC: out(512,2) = h2last(512,1024) @ fc_w(2,1024)^T + fc_b
__global__ void k_fc(const float* __restrict__ h2, const float* __restrict__ fw,
                     const float* __restrict__ fb, float* __restrict__ out) {
  int idx = blockIdx.x * blockDim.x + threadIdx.x;
  if (idx >= Bx * 2) return;
  int b = idx >> 1, o = idx & 1;
  const float* hv = h2 + (size_t)b * (2 * Hh);
  const float* wv = fw + (size_t)o * (2 * Hh);
  float s = fb[o];
  for (int k = 0; k < 2 * Hh; ++k) s += hv[k] * wv[k];
  out[idx] = s;
}

extern "C" void kernel_launch(void* const* d_in, const int* in_sizes, int n_in,
                              void* d_out, int out_size, void* d_ws, size_t ws_size,
                              hipStream_t stream) {
  (void)in_sizes; (void)n_in; (void)out_size; (void)ws_size;
  const int*   tokens = (const int*)  d_in[0];
  const float* emb    = (const float*)d_in[1];
  const float* w_ih0  = (const float*)d_in[2];
  const float* w_hh0  = (const float*)d_in[3];
  const float* b_ih0  = (const float*)d_in[4];
  const float* b_hh0  = (const float*)d_in[5];
  const float* w_ih1  = (const float*)d_in[6];
  const float* w_hh1  = (const float*)d_in[7];
  const float* b_ih1  = (const float*)d_in[8];
  const float* b_hh1  = (const float*)d_in[9];
  const float* fc_w   = (const float*)d_in[10];
  const float* fc_b   = (const float*)d_in[11];
  float* out = (float*)d_out;

  char* base = (char*)d_ws;
  size_t off = 0;
  auto take = [&](size_t bytes) -> char* {
    char* p = base + off;
    off = (off + bytes + 255) & ~(size_t)255;
    return p;
  };
  bf16*  xbf   = (bf16*) take((size_t)Mm * Ep * 2);
  bf16*  wih0b = (bf16*) take((size_t)NG * Ep * 2);
  bf16*  whh0b = (bf16*) take((size_t)NG * Hh * 2);
  bf16*  wih1b = (bf16*) take((size_t)NG * (2 * Hh) * 2);
  bf16*  whh1b = (bf16*) take((size_t)NG * Hh * 2);
  bf16*  gx    = (bf16*) take((size_t)Mm * NG * 2);      // reused: layer0 then layer1
  bf16*  h1seq = (bf16*) take((size_t)Mm * (2 * Hh) * 2);
  float* hfA   = (float*)take((size_t)2 * Bx * Hh * 4);
  float* hfB   = (float*)take((size_t)2 * Bx * Hh * 4);
  bf16*  hbA   = (bf16*) take((size_t)2 * Bx * Hh * 2);
  bf16*  hbB   = (bf16*) take((size_t)2 * Bx * Hh * 2);
  float* h2l   = (float*)take((size_t)Bx * (2 * Hh) * 4);

  // Weight conversion + embedding gather
  k_cast_pad<<<NG, 256, 0, stream>>>(w_ih0, wih0b, Ee, Ep);
  k_cast_pad<<<NG, 256, 0, stream>>>(w_hh0, whh0b, Hh, Hh);
  k_cast_pad<<<NG, 256, 0, stream>>>(w_ih1, wih1b, 2 * Hh, 2 * Hh);
  k_cast_pad<<<NG, 256, 0, stream>>>(w_hh1, whh1b, Hh, Hh);
  k_embed<<<Mm, 256, 0, stream>>>(tokens, emb, xbf);

  // Layer 0: gx = X @ Wih0^T (both directions, N=3072, K=320)
  k_gemm_bf16<<<dim3(Mm / BM, NG / BN), 256, 0, stream>>>(xbf, wih0b, gx, Ep, NG);

  (void)hipMemsetAsync(hfA, 0, (size_t)2 * Bx * Hh * 4, stream);
  (void)hipMemsetAsync(hfB, 0, (size_t)2 * Bx * Hh * 4, stream);
  (void)hipMemsetAsync(hbA, 0, (size_t)2 * Bx * Hh * 2, stream);
  (void)hipMemsetAsync(hbB, 0, (size_t)2 * Bx * Hh * 2, stream);

  float *hfc = hfA, *hfn = hfB;
  bf16  *hbc = hbA, *hbn = hbB;
  for (int s = 0; s < Tt; ++s) {
    k_gru_step<<<dim3(Bx / 32, Hh / 64, 2), 256, 0, stream>>>(
        hbc, hfc, whh0b, gx, b_ih0, b_hh0, hfn, hbn, h1seq, nullptr, s, 0, 0);
    float* tf = hfc; hfc = hfn; hfn = tf;
    bf16*  tb = hbc; hbc = hbn; hbn = tb;
  }

  // Layer 1: gx = h1seq @ Wih1^T (both directions, K=1024)
  k_gemm_bf16<<<dim3(Mm / BM, NG / BN), 256, 0, stream>>>(h1seq, wih1b, gx, 2 * Hh, NG);

  (void)hipMemsetAsync(hfA, 0, (size_t)2 * Bx * Hh * 4, stream);
  (void)hipMemsetAsync(hfB, 0, (size_t)2 * Bx * Hh * 4, stream);
  (void)hipMemsetAsync(hbA, 0, (size_t)2 * Bx * Hh * 2, stream);
  (void)hipMemsetAsync(hbB, 0, (size_t)2 * Bx * Hh * 2, stream);

  hfc = hfA; hfn = hfB; hbc = hbA; hbn = hbB;
  // Layer-1 forward: full 64 steps; only the final h is needed (write_last on s==T-1)
  for (int s = 0; s < Tt; ++s) {
    k_gru_step<<<dim3(Bx / 32, Hh / 64, 1), 256, 0, stream>>>(
        hbc, hfc, whh1b, gx, b_ih1, b_hh1, hfn, hbn, nullptr, h2l, s, 0, s == Tt - 1);
    float* tf = hfc; hfc = hfn; hfn = tf;
    bf16*  tb = hbc; hbc = hbn; hbn = tb;
  }
  // Layer-1 backward: output at t=T-1 is the FIRST backward scan step (h_prev = 0).
  // dir-1 region of state buffers was zeroed and never written by the fwd loop.
  k_gru_step<<<dim3(Bx / 32, Hh / 64, 1), 256, 0, stream>>>(
      hbc, hfc, whh1b, gx, b_ih1, b_hh1, hfn, hbn, nullptr, h2l, 0, 1, 1);

  // Final FC
  k_fc<<<(Bx * 2 + 255) / 256, 256, 0, stream>>>(h2l, fc_w, fc_b, out);
}
